// GraphConditionEncoder_40664750359186
// MI455X (gfx1250) — matrix operand order repaired and checked by reading, not compiled, not run on hardware
//
#include <hip/hip_runtime.h>
#include <hip/hip_bf16.h>

// Problem constants (match reference)
#define N_NODES 20000
#define N_EDGES 320000
#define NGRAPH  64
#define IN_DIM  128
#define FDIM    64
#define HEADS   4
#define HFDIM   256          // HEADS * FDIM
#define LATDIM  256
#define AGG_IN  832          // FDIM * (1 + 3*HEADS)

typedef __attribute__((ext_vector_type(16))) __bf16 v16bf;
typedef __attribute__((ext_vector_type(8)))  __bf16 v8bf;
typedef __attribute__((ext_vector_type(8)))  float  v8f;

// ---------------------------------------------------------------------------
// float atomic max via int-ordering trick (buffer always holds a valid float;
// init pattern must be 0xFF800000 == -inf)
__device__ __forceinline__ void atomicMaxF(float* addr, float v) {
  if (v >= 0.0f) atomicMax((int*)addr, __float_as_int(v));
  else           atomicMin((unsigned int*)addr, __float_as_uint(v));
}

// ---------------------------------------------------------------------------
__global__ void fill_u32_kernel(unsigned int* __restrict__ p, unsigned int v, int n) {
  int i = blockIdx.x * blockDim.x + threadIdx.x;
  if (i < n) p[i] = v;
}

__global__ void cvt_bf16_kernel(const float* __restrict__ in, __bf16* __restrict__ out, int n) {
  int i = blockIdx.x * blockDim.x + threadIdx.x;
  if (i < n) out[i] = (__bf16)in[i];
}

// ---------------------------------------------------------------------------
// Y[nrows, OUT] = X[nrows, K] @ W[OUT, K]^T   (bf16 in, fp32 accumulate)
//
// Block = 8 waves (256 threads: fills a WGP's 4 SIMD32s twice). Each wave owns
// a 16-row M-tile and all OUT columns (OUT/16 v8f accumulators in registers).
//
// K is processed in 32-wide stages. Per stage:
//   - weight chunk W[:, k0..k0+32) is staged in LDS, double buffered: global
//     loads for stage s+1 issue before the WMMAs of stage s (one barrier per
//     stage; the pre-store barrier is unnecessary with two buffers),
//   - the A fragment is double buffered in registers: stage s+1 activation
//     loads issue at the top of stage s's compute so WMMAs hide their latency,
//   - B fragments stream from LDS with a one-tile software prefetch, so the
//     inner loop waits on short dscnt counts only.
//
// LDS rows padded to 40 bf16 (80 B) so the 16 low lanes hit 16 distinct banks.
//
// Fragment layouts (A/B: contiguous k = 16*half + s, this code's choice, not the ISA table; D per ISA):
//  A (16-bit 16x32): lanes 0-15 = rows; lane half picks K {0..15} vs
//    {16..31} -> two aligned 16B loads from the bf16 activation row.
//  B (16-bit 32x16): lane n = column, lane half picks K {0..15} vs {16..31}
//    -> one contiguous 32B read of the (row-major) weight row.
//  D: VGPR r -> (M = r + hi*8, N = lane&15).
template<int K, int OUT>
__global__ __launch_bounds__(256)
void gemm_bf16_wmma(const __bf16* __restrict__ X, const __bf16* __restrict__ W,
                    float* __restrict__ Y, int nrows) {
  constexpr int NWAVES    = 8;
  constexpr int BLOCK     = 32 * NWAVES;
  constexpr int NT        = OUT / 16;        // N-tiles per wave
  constexpr int KC        = 32;              // K per LDS stage
  constexpr int NSTAGE    = K / KC;
  constexpr int LDSSTRIDE = KC + 8;          // pad: 40 bf16 = 80 B row stride
  constexpr int VPT       = (OUT * 4) / BLOCK; // v8bf staging vectors/thread

  __shared__ alignas(16) __bf16 wtile[2][OUT * LDSSTRIDE];

  const int tid   = threadIdx.x;
  const int wave  = tid >> 5;
  const int lane  = tid & 31;
  const int mbase = (blockIdx.x * NWAVES + wave) * 16;
  const bool active = (mbase < nrows);       // wave-uniform
  const int hi   = lane >> 4;
  const int nlo  = lane & 15;
  const int mrow = mbase + nlo;

  v8f zero = {};
  v8f acc[NT];
#pragma unroll
  for (int t = 0; t < NT; ++t) acc[t] = zero;

  const __bf16* xrow = X + (size_t)(active ? mrow : 0) * K;

  // ---- prologue: stage 0 of W into LDS buffer 0; A fragment for stage 0
  v8bf stage[VPT];
#pragma unroll
  for (int i = 0; i < VPT; ++i) {
    int v = tid + i * BLOCK, row = v >> 2, seg = v & 3;
    stage[i] = *(const v8bf*)(W + (size_t)row * K + seg * 8);
    *(v8bf*)(&wtile[0][row * LDSSTRIDE + seg * 8]) = stage[i];
  }
  v8bf alo = {}, ahi = {}, alo_n = {}, ahi_n = {};
  if (active) {
    alo = *(const v8bf*)(xrow + hi * 16);
    ahi = *(const v8bf*)(xrow + 8 + hi * 16);
  }
  __syncthreads();

  for (int s = 0; s < NSTAGE; ++s) {
    // issue global loads for weight stage s+1 (overlap with WMMAs below)
    if (s + 1 < NSTAGE) {
#pragma unroll
      for (int i = 0; i < VPT; ++i) {
        int v = tid + i * BLOCK, row = v >> 2, seg = v & 3;
        stage[i] = *(const v8bf*)(W + (size_t)row * K + (s + 1) * KC + seg * 8);
      }
    }
    if (active) {
      // issue next stage's A-fragment loads first so they hide under WMMAs
      if (s + 1 < NSTAGE) {
        const int kn = (s + 1) * KC;
        alo_n = *(const v8bf*)(xrow + kn + hi * 16);
        ahi_n = *(const v8bf*)(xrow + kn + 8 + hi * 16);
      }
      v16bf afrag;
#pragma unroll
      for (int i = 0; i < 8; ++i) { afrag[i] = alo[i]; afrag[8 + i] = ahi[i]; }

      const __bf16* bbase = &wtile[s & 1][0] + nlo * LDSSTRIDE + hi * 16;
      v16bf bcur = *(const v16bf*)(bbase);
#pragma unroll
      for (int t = 0; t < NT; ++t) {
        v16bf bnext = bcur;
        if (t + 1 < NT) bnext = *(const v16bf*)(bbase + (t + 1) * 16 * LDSSTRIDE);
        acc[t] = __builtin_amdgcn_wmma_f32_16x16x32_bf16(
            false, afrag, false, bcur, (short)0, acc[t], false, false);
        bcur = bnext;
      }
    }
    // Double buffered: readers of buffer (s+1)&1 finished before the previous
    // stage's barrier, so we can store immediately; one barrier per stage.
    if (s + 1 < NSTAGE) {
#pragma unroll
      for (int i = 0; i < VPT; ++i) {
        int v = tid + i * BLOCK, row = v >> 2, seg = v & 3;
        *(v8bf*)(&wtile[(s + 1) & 1][row * LDSSTRIDE + seg * 8]) = stage[i];
      }
      __syncthreads();
    }
    alo = alo_n;
    ahi = ahi_n;
  }

  if (active) {
#pragma unroll
    for (int t = 0; t < NT; ++t) {
#pragma unroll
      for (int r = 0; r < 8; ++r) {
        Y[(size_t)(mbase + hi * 8 + r) * OUT + t * 16 + nlo] = acc[t][r];
      }
    }
  }
}

// ---------------------------------------------------------------------------
// as[n,h] = xp[n,h,:] . a_src[h,:] ; ad likewise
__global__ void alphas_kernel(const float* __restrict__ xp,
                              const float* __restrict__ a_src,
                              const float* __restrict__ a_dst,
                              float* __restrict__ as_, float* __restrict__ ad_, int n) {
  int i = blockIdx.x * blockDim.x + threadIdx.x;   // over n*HEADS
  if (i >= n * HEADS) return;
  int node = i >> 2, h = i & 3;
  const float* xr = xp + (size_t)node * HFDIM + h * FDIM;
  const float* s  = a_src + h * FDIM;
  const float* d  = a_dst + h * FDIM;
  float vs = 0.f, vd = 0.f;
#pragma unroll 8
  for (int f = 0; f < FDIM; ++f) { float x = xr[f]; vs += x * s[f]; vd += x * d[f]; }
  as_[i] = vs;
  ad_[i] = vd;
}

__device__ __forceinline__ void edge_decode(int e, int E_,
                                            const int* __restrict__ srcI,
                                            const int* __restrict__ dstI,
                                            int& s, int& d) {
  if (e < E_) { s = srcI[e]; d = dstI[e]; }
  else        { s = e - E_;  d = e - E_; }          // appended self-loop
}

__global__ void edge_max_kernel(const int* __restrict__ srcI, const int* __restrict__ dstI,
                                int E_, int Ntot,
                                const float* __restrict__ as_, const float* __restrict__ ad_,
                                float* __restrict__ emax) {
  int e = blockIdx.x * blockDim.x + threadIdx.x;
  if (e >= E_ + Ntot) return;
  int s, d; edge_decode(e, E_, srcI, dstI, s, d);
#pragma unroll
  for (int h = 0; h < HEADS; ++h) {
    float v = as_[s * HEADS + h] + ad_[d * HEADS + h];
    v = v > 0.f ? v : 0.2f * v;                     // leaky_relu(.,0.2)
    atomicMaxF(&emax[d * HEADS + h], v);
  }
}

__global__ void edge_exp_kernel(const int* __restrict__ srcI, const int* __restrict__ dstI,
                                int E_, int Ntot,
                                const float* __restrict__ as_, const float* __restrict__ ad_,
                                const float* __restrict__ emax,
                                float* __restrict__ expw, float* __restrict__ denom) {
  int e = blockIdx.x * blockDim.x + threadIdx.x;
  if (e >= E_ + Ntot) return;
  int s, d; edge_decode(e, E_, srcI, dstI, s, d);
#pragma unroll
  for (int h = 0; h < HEADS; ++h) {
    float v = as_[s * HEADS + h] + ad_[d * HEADS + h];
    v = v > 0.f ? v : 0.2f * v;
    float w = __expf(v - emax[d * HEADS + h]);
    expw[(size_t)e * HEADS + h] = w;
    atomicAdd(&denom[d * HEADS + h], w);
  }
}

// out[dst] += expw[e,h] * xp[src]   (unnormalized; divide by denom at finalize)
// one thread per (edge, 4 consecutive features)
__global__ void edge_msg_kernel(const int* __restrict__ srcI, const int* __restrict__ dstI,
                                int E_, int Ntot,
                                const float* __restrict__ xp, const float* __restrict__ expw,
                                float* __restrict__ outAcc) {
  long t = (long)blockIdx.x * blockDim.x + threadIdx.x;
  long ET = (long)E_ + Ntot;
  if (t >= ET * 64) return;
  int e = (int)(t >> 6);
  int q = (int)(t & 63);                            // feature group: 4 floats
  int s, d; edge_decode(e, E_, srcI, dstI, s, d);
  int h = q >> 4;                                   // (q*4)/64
  float w = expw[(size_t)e * HEADS + h];
  // prefetch a future edge's source row (gfx1250 global_prefetch path)
  long e2 = (long)e + 2048;
  if (e2 < ET) {
    int s2 = (e2 < E_) ? srcI[e2] : (int)(e2 - E_);
    __builtin_prefetch(xp + (size_t)s2 * HFDIM + q * 4, 0, 1);
  }
  const float4 x4 = *(const float4*)(xp + (size_t)s * HFDIM + q * 4);
  float* o = outAcc + (size_t)d * HFDIM + q * 4;
  atomicAdd(o + 0, x4.x * w);
  atomicAdd(o + 1, x4.y * w);
  atomicAdd(o + 2, x4.z * w);
  atomicAdd(o + 3, x4.w * w);
}

// v = relu(outAcc/(denom+eps) + bias) ; write bf16 next-layer input ; pool max
__global__ void node_finalize_kernel(const float* __restrict__ outAcc,
                                     const float* __restrict__ denom,
                                     const float* __restrict__ bias,
                                     const int* __restrict__ batch,
                                     __bf16* __restrict__ xout,
                                     float* __restrict__ gbuf, int goff, int n) {
  int i = blockIdx.x * blockDim.x + threadIdx.x;    // over n*HFDIM
  if (i >= n * HFDIM) return;
  int node = i >> 8, hf = i & 255, h = hf >> 6;
  float v = outAcc[i] / (denom[node * HEADS + h] + 1e-16f) + bias[hf];
  v = fmaxf(v, 0.0f);
  xout[i] = (__bf16)v;
  atomicMaxF(&gbuf[(size_t)batch[node] * AGG_IN + goff + hf], v);
}

// fc epilogue: relu(Y+b), bf16 out, pool into g0 columns
__global__ void fc_finalize_kernel(const float* __restrict__ Y, const float* __restrict__ b,
                                   const int* __restrict__ batch,
                                   __bf16* __restrict__ xout, float* __restrict__ gbuf, int n) {
  int i = blockIdx.x * blockDim.x + threadIdx.x;    // over n*FDIM
  if (i >= n * FDIM) return;
  int node = i >> 6, f = i & 63;
  float v = fmaxf(Y[i] + b[f], 0.0f);
  xout[i] = (__bf16)v;
  atomicMaxF(&gbuf[(size_t)batch[node] * AGG_IN + f], v);
}

// out[g,l] = gbuf[g,:] . W_agg[l,:] + b_agg[l]   (64x256 outputs, tiny)
__global__ void final_gemm_kernel(const float* __restrict__ gbuf,
                                  const float* __restrict__ Wagg,
                                  const float* __restrict__ bagg,
                                  float* __restrict__ out) {
  int i = blockIdx.x * blockDim.x + threadIdx.x;
  if (i >= NGRAPH * LATDIM) return;
  int g = i >> 8, l = i & 255;
  const float* gr = gbuf + (size_t)g * AGG_IN;
  const float* wr = Wagg + (size_t)l * AGG_IN;
  float acc = bagg[l];
#pragma unroll 8
  for (int k = 0; k < AGG_IN; ++k) acc += gr[k] * wr[k];
  out[i] = acc;
}

// ---------------------------------------------------------------------------
extern "C" void kernel_launch(void* const* d_in, const int* in_sizes, int n_in,
                              void* d_out, int out_size, void* d_ws, size_t ws_size,
                              hipStream_t stream) {
  (void)in_sizes; (void)n_in; (void)out_size; (void)ws_size;
  const float* street = (const float*)d_in[0];
  const int*   eidx   = (const int*)d_in[1];
  const int*   batch  = (const int*)d_in[2];
  const float* W_fc   = (const float*)d_in[3];
  const float* b_fc   = (const float*)d_in[4];
  const float* W1     = (const float*)d_in[5];
  const float* a1s    = (const float*)d_in[6];
  const float* a1d    = (const float*)d_in[7];
  const float* b1     = (const float*)d_in[8];
  const float* W2     = (const float*)d_in[9];
  const float* a2s    = (const float*)d_in[10];
  const float* a2d    = (const float*)d_in[11];
  const float* b2     = (const float*)d_in[12];
  const float* W3     = (const float*)d_in[13];
  const float* a3s    = (const float*)d_in[14];
  const float* a3d    = (const float*)d_in[15];
  const float* b3     = (const float*)d_in[16];
  const float* W_agg  = (const float*)d_in[17];
  const float* b_agg  = (const float*)d_in[18];

  const int* srcI = eidx;
  const int* dstI = eidx + N_EDGES;

  // workspace carve-out (all 256B aligned)
  char* ws = (char*)d_ws;
  size_t off = 0;
  auto take = [&](size_t bytes) -> char* {
    char* p = ws + off;
    off = (off + bytes + 255) & ~(size_t)255;
    return p;
  };
  __bf16* xbf   = (__bf16*)take((size_t)N_NODES * HFDIM * 2);   // layer input (bf16)
  __bf16* wfcB  = (__bf16*)take((size_t)FDIM * IN_DIM * 2);
  __bf16* w1B   = (__bf16*)take((size_t)HFDIM * FDIM * 2);
  __bf16* w2B   = (__bf16*)take((size_t)HFDIM * HFDIM * 2);
  __bf16* w3B   = (__bf16*)take((size_t)HFDIM * HFDIM * 2);
  float*  xp    = (float*)take((size_t)N_NODES * HFDIM * 4);    // GEMM output / messages
  float*  outAcc= (float*)take((size_t)N_NODES * HFDIM * 4);    // scatter accumulator
  float*  as_   = (float*)take((size_t)N_NODES * HEADS * 4);
  float*  ad_   = (float*)take((size_t)N_NODES * HEADS * 4);
  float*  emax  = (float*)take((size_t)N_NODES * HEADS * 4);
  float*  denom = (float*)take((size_t)N_NODES * HEADS * 4);
  float*  expw  = (float*)take((size_t)(N_EDGES + N_NODES) * HEADS * 4);
  float*  gbuf  = (float*)take((size_t)NGRAPH * AGG_IN * 4);

  const int TB = 256;
  auto nb = [](long n, int tb) { return (int)((n + tb - 1) / tb); };
  const int ET = N_EDGES + N_NODES;
  const int gemmGrid = (N_NODES + 127) / 128;                   // 8 waves x 16 rows

  // bf16 conversions + pooling buffer init (-inf bit pattern)
  cvt_bf16_kernel<<<nb((long)N_NODES * IN_DIM, TB), TB, 0, stream>>>(street, xbf, N_NODES * IN_DIM);
  cvt_bf16_kernel<<<nb(FDIM * IN_DIM, TB), TB, 0, stream>>>(W_fc, wfcB, FDIM * IN_DIM);
  cvt_bf16_kernel<<<nb(HFDIM * FDIM, TB), TB, 0, stream>>>(W1, w1B, HFDIM * FDIM);
  cvt_bf16_kernel<<<nb(HFDIM * HFDIM, TB), TB, 0, stream>>>(W2, w2B, HFDIM * HFDIM);
  cvt_bf16_kernel<<<nb(HFDIM * HFDIM, TB), TB, 0, stream>>>(W3, w3B, HFDIM * HFDIM);
  fill_u32_kernel<<<nb(NGRAPH * AGG_IN, TB), TB, 0, stream>>>((unsigned*)gbuf, 0xFF800000u, NGRAPH * AGG_IN);

  // fc: n0 = relu(x @ W_fc^T + b_fc)  -> bf16 into xbf, pool g0
  gemm_bf16_wmma<IN_DIM, FDIM><<<gemmGrid, 256, 0, stream>>>(xbf, wfcB, xp, N_NODES);
  fc_finalize_kernel<<<nb((long)N_NODES * FDIM, TB), TB, 0, stream>>>(xp, b_fc, batch, xbf, gbuf, N_NODES);

  auto run_gat = [&](int layer, const __bf16* winB, const float* asrc, const float* adst,
                     const float* bias, int goff) {
    if (layer == 1)
      gemm_bf16_wmma<FDIM, HFDIM><<<gemmGrid, 256, 0, stream>>>(xbf, winB, xp, N_NODES);
    else
      gemm_bf16_wmma<HFDIM, HFDIM><<<gemmGrid, 256, 0, stream>>>(xbf, winB, xp, N_NODES);
    alphas_kernel<<<nb((long)N_NODES * HEADS, TB), TB, 0, stream>>>(xp, asrc, adst, as_, ad_, N_NODES);
    fill_u32_kernel<<<nb((long)N_NODES * HEADS, TB), TB, 0, stream>>>((unsigned*)emax, 0xFF800000u, N_NODES * HEADS);
    fill_u32_kernel<<<nb((long)N_NODES * HEADS, TB), TB, 0, stream>>>((unsigned*)denom, 0u, N_NODES * HEADS);
    fill_u32_kernel<<<nb((long)N_NODES * HFDIM, TB), TB, 0, stream>>>((unsigned*)outAcc, 0u, N_NODES * HFDIM);
    edge_max_kernel<<<nb(ET, TB), TB, 0, stream>>>(srcI, dstI, N_EDGES, N_NODES, as_, ad_, emax);
    edge_exp_kernel<<<nb(ET, TB), TB, 0, stream>>>(srcI, dstI, N_EDGES, N_NODES, as_, ad_, emax, expw, denom);
    edge_msg_kernel<<<nb((long)ET * 64, TB), TB, 0, stream>>>(srcI, dstI, N_EDGES, N_NODES, xp, expw, outAcc);
    node_finalize_kernel<<<nb((long)N_NODES * HFDIM, TB), TB, 0, stream>>>(outAcc, denom, bias, batch, xbf, gbuf, goff, N_NODES);
  };

  run_gat(1, w1B, a1s, a1d, b1, FDIM);                 // g1 cols [64,320)
  run_gat(2, w2B, a2s, a2d, b2, FDIM + HFDIM);         // g2 cols [320,576)
  run_gat(3, w3B, a3s, a3d, b3, FDIM + 2 * HFDIM);     // g3 cols [576,832)

  final_gemm_kernel<<<nb(NGRAPH * LATDIM, TB), TB, 0, stream>>>(gbuf, W_agg, b_agg, (float*)d_out);
}
